// BiLSTM_CRF_51204600103386
// MI455X (gfx1250) — matrix operand order repaired and checked by reading, not compiled, not run on hardware
//
#include <hip/hip_runtime.h>
#include <hip/hip_bf16.h>

// ---------------- problem constants ----------------
#define V_  50000
#define E_  1024
#define H_  512
#define L_  2
#define C_  24
#define B_  32
#define T_  512
#define FOURH 2048           // 4*H
#define TB  (T_*B_)          // 16384 rows for the big GEMMs

// ---------------- vector types ----------------
typedef __attribute__((ext_vector_type(16))) __bf16 bf16x16;
typedef __attribute__((ext_vector_type(8)))  __bf16 bf16x8;
typedef __attribute__((ext_vector_type(8)))  float  f32x8;
typedef __attribute__((ext_vector_type(4)))  unsigned int u32x4;
typedef __attribute__((ext_vector_type(8)))  int    i32x8;
typedef __attribute__((ext_vector_type(4)))  int    i32x4;

#if __has_builtin(__builtin_amdgcn_tensor_load_to_lds)
#define HAVE_TDM 1
#endif

__device__ __forceinline__ f32x8 wmma_bf16(bf16x16 a, bf16x16 b, f32x8 c) {
  // D = A(16x32) * B(32x16) + C, f32 accumulate  -> v_wmma_f32_16x16x32_bf16
  return __builtin_amdgcn_wmma_f32_16x16x32_bf16(false, a, false, b, (short)0, c,
                                                 false, false);
}

// A fragment (16x32 bf16, row-major source, rows m..m+15, k-chunk kk*32).
// Lane l<16: row m+l, K 0..15; lane l>=16: row m+l-16, K 16..31.
__device__ __forceinline__ bf16x16 frag_a_g(const __bf16* __restrict__ base,
                                            int ldk, int m, int kk, int lane) {
  const __bf16* p = base + (size_t)(m + (lane & 15)) * ldk + (kk << 5) + ((lane >> 4) << 4);
  bf16x16 r;
  *reinterpret_cast<bf16x8*>(&r)       = *reinterpret_cast<const bf16x8*>(p);
  *(reinterpret_cast<bf16x8*>(&r) + 1) = *reinterpret_cast<const bf16x8*>(p + 8);
  return r;
}

// B fragment (32x16 bf16) from row-major W (X*W^T): lane l<16: col n+l, K 0..15
// contiguous; lane l>=16: col n+l-16, K 16..31.  One contiguous 32B load.
__device__ __forceinline__ bf16x16 frag_b_g(const __bf16* __restrict__ base,
                                            int ldk, int n, int kk, int lane) {
  const __bf16* p = base + (size_t)(n + (lane & 15)) * ldk + (kk << 5) + ((lane >> 4) << 4);
  return *reinterpret_cast<const bf16x16*>(p);
}

__device__ __forceinline__ float sigmf(float x) { return 1.f / (1.f + __expf(-x)); }

// ---------------- Tensor Data Mover: 2D tile (rows x kspan) of bf16 ----------
// D# per CDNA5 ISA ch.8: group0 = {count/flags, lds_addr, global_addr, type=2},
// group1 = {data_size/flags, dims, tile dims, dim0 stride}.  Groups 2/3 zero
// (2D tensor).  Tracked by TENSORcnt.
__device__ __forceinline__ void tdm_load_2d_bf16(unsigned lds_addr, const void* gsrc,
                                                 unsigned rows, unsigned kspan,
                                                 unsigned ldk) {
#if defined(HAVE_TDM)
  unsigned long long ga = (unsigned long long)(uintptr_t)gsrc;
  u32x4 g0 = {1u,                                   // count=1, user descriptor
              lds_addr,                              // LDS byte address
              (unsigned)ga,                          // global addr [31:0]
              (unsigned)((ga >> 32) & 0x1FFFFFFu) | (2u << 30)}; // [56:32] | type=2
  i32x8 g1;
  g1[0] = (int)(1u << 16);                 // workgroup_mask=0, data_size=1 (2B)
  g1[1] = (int)(ldk << 16);                // tensor_dim0[15:0] (K extent)
  g1[2] = (int)(4096u << 16);              // tensor_dim0 hi=0 | tensor_dim1 lo
  g1[3] = (int)(kspan << 16);              // tensor_dim1 hi=0 | tile_dim0 = kspan
  g1[4] = (int)rows;                       // tile_dim1 = rows, tile_dim2 = 0
  g1[5] = (int)ldk;                        // tensor_dim0_stride[31:0]
  g1[6] = 0;                               // stride0 hi | tensor_dim1_stride lo
  g1[7] = 0;
  i32x4 gz = {0, 0, 0, 0};
#if __clang_major__ >= 23
  i32x8 gz8 = {0, 0, 0, 0, 0, 0, 0, 0};
  __builtin_amdgcn_tensor_load_to_lds(g0, g1, gz, gz, gz8, 0);
#else
  __builtin_amdgcn_tensor_load_to_lds(g0, g1, gz, gz, 0);
#endif
#else
  (void)lds_addr; (void)gsrc; (void)rows; (void)kspan; (void)ldk;
#endif
}

// ---------------- small prep kernels ----------------
__global__ void cvt_bf16_kernel(const float* __restrict__ s, __bf16* __restrict__ d, size_t n) {
  size_t i = (size_t)blockIdx.x * blockDim.x + threadIdx.x;
  if (i < n) d[i] = (__bf16)s[i];
}

__global__ void bias_sum_kernel(const float* __restrict__ a, const float* __restrict__ b,
                                float* __restrict__ d, int n) {
  int i = blockIdx.x * blockDim.x + threadIdx.x;
  if (i < n) d[i] = a[i] + b[i];
}

// pad fc_w [24,1024] -> bf16 [32,1024] (rows 24..31 zero)
__global__ void fc_pad_kernel(const float* __restrict__ w, __bf16* __restrict__ d) {
  int i = blockIdx.x * blockDim.x + threadIdx.x;   // < 32*1024
  int r = i >> 10;
  d[i] = (r < C_) ? (__bf16)w[i] : (__bf16)0.f;
}

// embedding gather: xb[t][b][e] = bf16(emb[x[b][t]][e])
__global__ void embed_kernel(const int* __restrict__ ids, const float* __restrict__ emb,
                             __bf16* __restrict__ xb) {
  size_t i = (size_t)blockIdx.x * blockDim.x + threadIdx.x;  // < T*B*E
  int e = (int)(i % E_);
  size_t tb = i / E_;
  int b = (int)(tb % B_);
  int t = (int)(tb / B_);
  int id = ids[(size_t)b * T_ + t];
  xb[i] = (__bf16)emb[(size_t)id * E_ + e];
}

// ---------------- big input-projection GEMM ----------------
// G[m][n] = sum_k X[m][k]*W[n][k] + bias[n]; M=TB, N=FOURH, K=1024.
// Block = 256 thr = 8 waves, 128-row x 64-col macro-tile.  Wave w: rows
// [w*16, w*16+16), 4 accumulators over the 64-col panel (A-frag reused 4x).
// The 64x128k B panel is TDM-staged into LDS (double-buffered, DMA overlaps
// compute); A streams straight from global (pipelined b128 loads).
#define KSPAN 128                         // staged K chunk (4 wmma k-steps)
#define NSTAGE (1024 / KSPAN)             // 8
__global__ void __launch_bounds__(256)
gates_gemm_kernel(const __bf16* __restrict__ X, const __bf16* __restrict__ W,
                  const float* __restrict__ bias, float* __restrict__ G) {
  const int K = 1024;
  __shared__ __align__(32) __bf16 bsh[2][64 * KSPAN];   // 2 x 16 KB

  const int tid  = threadIdx.x;
  const int w    = tid >> 5;
  const int lane = tid & 31;
  const int nblk = FOURH / 64;                 // 32
  const int mb   = blockIdx.x / nblk;          // 0..127
  const int nb   = blockIdx.x % nblk;          // 0..31
  const int mrow = (mb << 7) + (w << 4);       // this wave's 16 rows

  const __bf16* Wp = W + (size_t)(nb * 64) * K;   // 64-row weight panel

  const f32x8 vzero = {0.f, 0.f, 0.f, 0.f, 0.f, 0.f, 0.f, 0.f};
  f32x8 acc[4];
#pragma unroll
  for (int nt = 0; nt < 4; ++nt) acc[nt] = vzero;

#if defined(HAVE_TDM)
  if (w == 0) {
    tdm_load_2d_bf16((unsigned)(uintptr_t)&bsh[0][0], Wp, 64, KSPAN, K);
    __builtin_amdgcn_s_wait_tensorcnt(0);
  }
  __syncthreads();
#endif

  for (int s = 0; s < NSTAGE; ++s) {
    const int cur = s & 1;
#if defined(HAVE_TDM)
    if (w == 0 && s + 1 < NSTAGE) {   // DMA next panel chunk into the free buffer
      tdm_load_2d_bf16((unsigned)(uintptr_t)&bsh[cur ^ 1][0],
                       (const char*)Wp + (size_t)(s + 1) * KSPAN * 2, 64, KSPAN, K);
    }
#else
    // fallback: cooperative synchronous staging
    __syncthreads();
    for (int i = tid; i < 64 * KSPAN / 8; i += 256) {
      int r8 = i >> 4;              // row (64 rows, 16 chunks of 8 elems)
      int c8 = i & 15;
      *reinterpret_cast<bf16x8*>(&bsh[cur][r8 * KSPAN + c8 * 8]) =
          *reinterpret_cast<const bf16x8*>(Wp + (size_t)r8 * K + s * KSPAN + c8 * 8);
    }
    __syncthreads();
#endif

#pragma unroll
    for (int k2 = 0; k2 < KSPAN / 32; ++k2) {
      const int kk = s * (KSPAN / 32) + k2;
      bf16x16 a = frag_a_g(X, K, mrow, kk, lane);
#pragma unroll
      for (int nt = 0; nt < 4; ++nt) {
        // B fragment from the LDS panel ([64 cols][KSPAN] row-major)
        const __bf16* bp = &bsh[cur][(size_t)((nt << 4) + (lane & 15)) * KSPAN +
                                     (k2 << 5) + ((lane >> 4) << 4)];
        bf16x16 b = *reinterpret_cast<const bf16x16*>(bp);
        acc[nt] = wmma_bf16(a, b, acc[nt]);
      }
    }

#if defined(HAVE_TDM)
    if (w == 0) __builtin_amdgcn_s_wait_tensorcnt(0);  // next panel landed
    __syncthreads();                                   // all done reading cur
#endif
  }

#pragma unroll
  for (int nt = 0; nt < 4; ++nt) {
    int n = (nb << 6) + (nt << 4) + (lane & 15);
    float bv = bias[n];
#pragma unroll
    for (int r = 0; r < 8; ++r) {
      int row = mrow + r + ((lane >> 4) << 3);
      G[(size_t)row * FOURH + n] = acc[nt][r] + bv;
    }
  }
}

// ---------------- persistent recurrent kernel ----------------
// grid = 2 blocks (dir), block = 1024 threads = 32 waves.
// Wave w owns h-columns [16w, 16w+16) for all 4 gate sections and both 16-row
// batch tiles; cell state c lives in registers across all T steps; h round-
// trips through LDS (bf16) for the next step's A fragments.
__global__ void __launch_bounds__(1024)
lstm_recurrent_kernel(const float* __restrict__ G,     // [2][T][B][4H] f32
                      const __bf16* __restrict__ Whh,  // [2][4H][H] bf16
                      __bf16* __restrict__ Y) {        // [T][B][2H] bf16
  __shared__ __align__(32) __bf16 h_sh[B_ * H_];       // 32 KB

  const int dir  = blockIdx.x;
  const int tid  = threadIdx.x;
  const int w    = tid >> 5;        // 0..31  == h-column tile
  const int lane = tid & 31;

  const float*  Gd   = G   + (size_t)dir * T_ * B_ * FOURH;
  const __bf16* Whhd = Whh + (size_t)dir * FOURH * H_;

  for (int i = tid; i < B_ * H_; i += 1024) h_sh[i] = (__bf16)0.f;
  __syncthreads();

  const f32x8 vzero = {0.f, 0.f, 0.f, 0.f, 0.f, 0.f, 0.f, 0.f};
  f32x8 cst[2];
  cst[0] = vzero;
  cst[1] = vzero;

  const int hcol = (w << 4) + (lane & 15);

  for (int t = 0; t < T_; ++t) {
    const int tt = dir ? (T_ - 1 - t) : t;

    f32x8 acc[2][4];
#pragma unroll
    for (int m = 0; m < 2; ++m)
#pragma unroll
      for (int s = 0; s < 4; ++s) acc[m][s] = vzero;

    // R = h_prev @ Whh^T  (K = 512)
#pragma unroll 2
    for (int kk = 0; kk < H_ / 32; ++kk) {
      bf16x16 a[2];
#pragma unroll
      for (int m = 0; m < 2; ++m) {
        const __bf16* ap = &h_sh[(size_t)((m << 4) + (lane & 15)) * H_ +
                                 (kk << 5) + ((lane >> 4) << 4)];
        *reinterpret_cast<bf16x8*>(&a[m])       = *reinterpret_cast<const bf16x8*>(ap);
        *(reinterpret_cast<bf16x8*>(&a[m]) + 1) = *reinterpret_cast<const bf16x8*>(ap + 8);
      }
#pragma unroll
      for (int s = 0; s < 4; ++s) {
        bf16x16 b = frag_b_g(Whhd, H_, s * H_ + (w << 4), kk, lane);
        acc[0][s] = wmma_bf16(a[0], b, acc[0][s]);
        acc[1][s] = wmma_bf16(a[1], b, acc[1][s]);
      }
    }

    __syncthreads();  // everyone done reading old h_sh

    // gate fusion: i,f,g,o -> c,h ; write h to LDS + layer output
#pragma unroll
    for (int m = 0; m < 2; ++m) {
#pragma unroll
      for (int r = 0; r < 8; ++r) {
        int brow = (m << 4) + r + ((lane >> 4) << 3);
        const float* gp = Gd + ((size_t)tt * B_ + brow) * FOURH;
        float gi = acc[m][0][r] + gp[hcol];
        float gf = acc[m][1][r] + gp[H_ + hcol];
        float gg = acc[m][2][r] + gp[2 * H_ + hcol];
        float go = acc[m][3][r] + gp[3 * H_ + hcol];
        float iv = sigmf(gi);
        float fv = sigmf(gf);
        float gv = tanhf(gg);
        float ov = sigmf(go);
        float cv = fv * cst[m][r] + iv * gv;
        cst[m][r] = cv;
        float hv = ov * tanhf(cv);
        __bf16 hb = (__bf16)hv;
        h_sh[(size_t)brow * H_ + hcol] = hb;
        Y[((size_t)tt * B_ + brow) * (2 * H_) + dir * H_ + hcol] = hb;
      }
    }
    __syncthreads();  // h_sh fully updated for next step
  }
}

// ---------------- FC layer (WMMA, N padded 24->32) ----------------
__global__ void fc_wmma_kernel(const __bf16* __restrict__ X,   // [TB,1024]
                               const __bf16* __restrict__ W,   // [32,1024] padded
                               const float* __restrict__ bias, // [24]
                               float* __restrict__ out) {      // [B][T][C]
  const int K = 1024;
  int wid  = (blockIdx.x * blockDim.x + threadIdx.x) >> 5;
  int lane = threadIdx.x & 31;
  int mt = wid >> 1;
  int nt = wid & 1;
  f32x8 acc = {0.f, 0.f, 0.f, 0.f, 0.f, 0.f, 0.f, 0.f};
#pragma unroll 4
  for (int kk = 0; kk < K / 32; ++kk) {
    bf16x16 a = frag_a_g(X, K, mt << 4, kk, lane);
    bf16x16 b = frag_b_g(W, K, nt << 4, kk, lane);
    acc = wmma_bf16(a, b, acc);
  }
  int c = (nt << 4) + (lane & 15);
  if (c < C_) {
    float bv = bias[c];
#pragma unroll
    for (int r = 0; r < 8; ++r) {
      int m = (mt << 4) + r + ((lane >> 4) << 3);
      int t = m >> 5;       // B_ == 32
      int b = m & 31;
      out[((size_t)b * T_ + t) * C_ + c] = acc[r] + bv;
    }
  }
}

// ---------------- CRF log-likelihood ----------------
__global__ void crf_kernel(const float* __restrict__ logits,   // [B][T][C]
                           const int* __restrict__ labels,     // [B][T]
                           const float* __restrict__ start,
                           const float* __restrict__ endv,
                           const float* __restrict__ trans,    // [C][C]
                           float* __restrict__ out_neg_llh) {
  __shared__ float al[B_][C_];
  __shared__ float tr[C_][C_];
  __shared__ float red[B_];
  int tid = threadIdx.x;
  int b = tid / C_;
  int c = tid % C_;
  if (tid < C_ * C_) tr[tid / C_][tid % C_] = trans[tid];
  __syncthreads();

  al[b][c] = start[c] + logits[((size_t)b * T_ + 0) * C_ + c];
  __syncthreads();

  for (int t = 1; t < T_; ++t) {
    float mx = -1e30f;
#pragma unroll 4
    for (int p = 0; p < C_; ++p) mx = fmaxf(mx, al[b][p] + tr[p][c]);
    float s = 0.f;
#pragma unroll 4
    for (int p = 0; p < C_; ++p) s += __expf(al[b][p] + tr[p][c] - mx);
    float na = mx + __logf(s) + logits[((size_t)b * T_ + t) * C_ + c];
    __syncthreads();
    al[b][c] = na;
    __syncthreads();
  }

  if (c == 0) {
    float mx = -1e30f;
    for (int p = 0; p < C_; ++p) mx = fmaxf(mx, al[b][p] + endv[p]);
    float s = 0.f;
    for (int p = 0; p < C_; ++p) s += __expf(al[b][p] + endv[p] - mx);
    float logZ = mx + __logf(s);

    const int* lab = labels + (size_t)b * T_;
    float num = start[lab[0]] + endv[lab[T_ - 1]];
    for (int t = 0; t < T_; ++t) num += logits[((size_t)b * T_ + t) * C_ + lab[t]];
    for (int t = 1; t < T_; ++t) num += tr[lab[t - 1]][lab[t]];
    red[b] = logZ - num;   // contribution to -llh
  }
  __syncthreads();
  if (tid == 0) {
    float s = 0.f;
    for (int i = 0; i < B_; ++i) s += red[i];
    out_neg_llh[0] = s;
  }
}

// ---------------- host-side orchestration ----------------
extern "C" void kernel_launch(void* const* d_in, const int* in_sizes, int n_in,
                              void* d_out, int out_size, void* d_ws, size_t ws_size,
                              hipStream_t stream) {
  (void)in_sizes; (void)n_in; (void)out_size; (void)ws_size;
  const int*   x      = (const int*)d_in[0];
  const int*   labels = (const int*)d_in[1];
  const float* emb    = (const float*)d_in[2];
  const float* w_ih   = (const float*)d_in[3];
  const float* w_hh   = (const float*)d_in[4];
  const float* b_ih   = (const float*)d_in[5];
  const float* b_hh   = (const float*)d_in[6];
  const float* fc_w   = (const float*)d_in[7];
  const float* fc_b   = (const float*)d_in[8];
  const float* crf_s  = (const float*)d_in[9];
  const float* crf_e  = (const float*)d_in[10];
  const float* crf_t  = (const float*)d_in[11];
  float* out = (float*)d_out;

  // workspace carve-up (256B aligned)
  char* ws = (char*)d_ws;
  size_t off = 0;
  auto carve = [&](size_t bytes) {
    void* p = ws + off;
    off += (bytes + 255) & ~(size_t)255;
    return p;
  };
  __bf16* xb    = (__bf16*)carve((size_t)TB * E_ * 2);            // [T,B,E]
  __bf16* y0    = (__bf16*)carve((size_t)TB * 1024 * 2);          // layer-1 out
  __bf16* y1    = (__bf16*)carve((size_t)TB * 1024 * 2);          // layer-2 out
  __bf16* wihb  = (__bf16*)carve((size_t)L_ * 2 * FOURH * E_ * 2);
  __bf16* whhb  = (__bf16*)carve((size_t)L_ * 2 * FOURH * H_ * 2);
  __bf16* fcwb  = (__bf16*)carve((size_t)32 * 1024 * 2);
  float*  biasS = (float*)carve((size_t)L_ * 2 * FOURH * 4);
  float*  gates = (float*)carve((size_t)2 * T_ * B_ * FOURH * 4); // both dirs, 1 layer

  const int THR = 256;
  {
    size_t n = (size_t)L_ * 2 * FOURH * E_;                       // 8,388,608
    cvt_bf16_kernel<<<(int)((n + THR - 1) / THR), THR, 0, stream>>>(w_ih, wihb, n);
  }
  {
    size_t n = (size_t)L_ * 2 * FOURH * H_;                       // 4,194,304
    cvt_bf16_kernel<<<(int)((n + THR - 1) / THR), THR, 0, stream>>>(w_hh, whhb, n);
  }
  bias_sum_kernel<<<(L_ * 2 * FOURH + THR - 1) / THR, THR, 0, stream>>>(
      b_ih, b_hh, biasS, L_ * 2 * FOURH);
  fc_pad_kernel<<<(32 * 1024) / THR, THR, 0, stream>>>(fc_w, fcwb);

  {
    size_t n = (size_t)TB * E_;                                   // 16,777,216
    embed_kernel<<<(int)(n / THR), THR, 0, stream>>>(x, emb, xb);
  }

  // two stacked bidirectional layers
  const int gemm_blocks = (TB / 128) * (FOURH / 64);              // 4096
  for (int l = 0; l < L_; ++l) {
    const __bf16* Xin = (l == 0) ? xb : y0;
    __bf16*       Yl  = (l == 0) ? y0 : y1;
    for (int d = 0; d < 2; ++d) {
      gates_gemm_kernel<<<gemm_blocks, THR, 0, stream>>>(
          Xin, wihb + (size_t)(l * 2 + d) * FOURH * E_,
          biasS + (size_t)(l * 2 + d) * FOURH,
          gates + (size_t)d * T_ * B_ * FOURH);
    }
    lstm_recurrent_kernel<<<2, 1024, 0, stream>>>(
        gates, whhb + (size_t)l * 2 * FOURH * H_, Yl);
  }

  // FC -> logits in d_out ([B,T,C])
  const int fc_blocks = (TB / 16) * 2 / (THR / 32);               // 256
  fc_wmma_kernel<<<fc_blocks, THR, 0, stream>>>(y1, fcwb, fc_b, out);

  // CRF -> -llh scalar at out[B*T*C]
  crf_kernel<<<1, B_ * C_, 0, stream>>>(out, labels, crf_s, crf_e, crf_t,
                                        out + (size_t)B_ * T_ * C_);
}
